// BotGAT_44856638439809
// MI455X (gfx1250) — matrix operand order repaired and checked by reading, not compiled, not run on hardware
//
#include <hip/hip_runtime.h>
#include <hip/hip_bf16.h>

// ---------------------------------------------------------------------------
// BotGAT forward for MI455X (gfx1250, wave32).
//  - Dense GEMMs: v_wmma_f32_16x16x32_f16, B staged global->LDS via TDM
//    (tensor_load_to_lds + s_wait_tensorcnt), read back as ds_load_b128.
//  - Edge softmax/aggregate: f32 atomics resolving in the 192MB L2.
// ---------------------------------------------------------------------------

typedef _Float16 v16h __attribute__((ext_vector_type(16)));
typedef _Float16 v8h  __attribute__((ext_vector_type(8)));
typedef float    v8f  __attribute__((ext_vector_type(8)));
typedef unsigned int u32x4 __attribute__((ext_vector_type(4)));
typedef int          i32x8 __attribute__((ext_vector_type(8)));
typedef int          i32x4 __attribute__((ext_vector_type(4)));

#if __has_builtin(__builtin_amdgcn_tensor_load_to_lds)
#define HAVE_TDM 1
#endif

#define LREL(x, s) ((x) > 0.0f ? (x) : (s) * (x))

#define KC 128   // K-chunk staged in LDS per iteration
#define TM 4     // M tiles (16 rows each) per wave

// ---------------------------------------------------------------------------
// C[M,Ncols](ldc) = act(A[M,K](lda) @ W[K,Ncols](ldw) + bias)
// block = (32,4): 4 waves x TM tiles = 256 rows x 16 cols per block.
// Fragment layouts per CDNA5 ISA 7.12.2 (wave32):
//   A 16x32 f16 : lane l: m=l%16, g=l/16; two contiguous 8-float runs:
//                 K = g*8..g*8+7  and  K = 16+g*8..16+g*8+7
//   B 32x16 f16 : lane l: n=l%16, g=l/16; K = 8g+0..7 and 16+8g+0..7 -> two 8-half runs in WT[n][k]
//   C 16x16 f32 : lane l: n=l%16, g=l/16; acc[v] -> M = v+8g
// ---------------------------------------------------------------------------
__global__ __launch_bounds__(128)
void gemm_wmma_f16(const float* __restrict__ A, int lda,
                   const float* __restrict__ W, int ldw,
                   const float* __restrict__ bias,
                   float* __restrict__ C, int ldc,
                   int M, int K, int Ncols,
                   float slope, int do_act)
{
    __shared__ __align__(16) float    Wst[KC * 16];  // staged f32 W chunk [k][n]
    __shared__ __align__(32) _Float16 WT[16 * KC];   // transposed f16    [n][k]

    const int lane = threadIdx.x;            // 0..31
    const int wid  = threadIdx.y;            // 0..3
    const int tid  = wid * 32 + lane;        // 0..127
    const int n0   = blockIdx.y * 16;
    const int tileBase = blockIdx.x * (4 * TM) + wid * TM;   // wave's first M tile

    const int g = lane >> 4;
    const int r = lane & 15;

    v8f acc[TM];
#pragma unroll
    for (int t = 0; t < TM; ++t) acc[t] = (v8f){};

    for (int k0 = 0; k0 < K; k0 += KC) {
        // ---- stage W[k0:k0+KC, n0:n0+16] into LDS ------------------------
#ifdef HAVE_TDM
        if (wid == 0) {
            // Tensor DMA descriptor (D#), 2D tile: dim0=16 cols, dim1=KC rows
            const unsigned long long ga =
                (unsigned long long)(uintptr_t)(W + (size_t)k0 * ldw + n0);
            const unsigned lds = (unsigned)(unsigned long long)(uintptr_t)Wst;
            u32x4 g0v;
            g0v[0] = 1u;                                   // count=1, no gather
            g0v[1] = lds;                                  // lds_addr
            g0v[2] = (unsigned)(ga & 0xFFFFFFFFu);         // global_addr[31:0]
            g0v[3] = (unsigned)((ga >> 32) & 0x01FFFFFFu)  // global_addr[56:32]
                     | (2u << 30);                         // type=2 (image)
            i32x8 g1v;
            g1v[0] = (int)(2u << 16);                      // data_size=4B
            g1v[1] = (int)((16u & 0xFFFFu) << 16);         // tensor_dim0 lo16
            g1v[2] = (int)(((unsigned)KC & 0xFFFFu) << 16);// dim0 hi | tensor_dim1 lo
            g1v[3] = (int)(16u << 16);                     // dim1 hi | tile_dim0=16
            g1v[4] = (int)((unsigned)KC & 0xFFFFu);        // tile_dim1=KC, tile_dim2=0
            g1v[5] = (int)(unsigned)ldw;                   // tensor_dim0_stride lo32
            g1v[6] = 0;
            g1v[7] = 0;
            i32x4 z4; z4[0] = 0; z4[1] = 0; z4[2] = 0; z4[3] = 0;
            i32x8 z8;
#pragma unroll
            for (int i = 0; i < 8; ++i) z8[i] = 0;
            // 6-arg toolchain variant: (g0, g1, g2, g3, extra, cpol)
            __builtin_amdgcn_tensor_load_to_lds(g0v, g1v, z4, z4, z8, 0);
            __builtin_amdgcn_s_wait_tensorcnt(0);
        }
#else
        for (int idx = tid; idx < KC * 16; idx += 128) {
            const int k = idx >> 4, n = idx & 15;
            Wst[idx] = W[(size_t)(k0 + k) * ldw + (n0 + n)];
        }
#endif
        __syncthreads();
        // convert + transpose to f16: WT[n][k]
        for (int idx = tid; idx < KC * 16; idx += 128) {
            const int k = idx >> 4, n = idx & 15;
            WT[n * KC + k] = (_Float16)Wst[idx];
        }
        __syncthreads();

        for (int kk = 0; kk < KC; kk += 32) {
            // B fragment from LDS: two 8-half runs at WT[r][kk + 8g] and WT[r][kk + 16 + 8g]
            const v8h blo = *(const v8h*)&WT[r * KC + kk + g * 8];
            const v8h bhi = *(const v8h*)&WT[r * KC + kk + 16 + g * 8];
            v16h b;
#pragma unroll
            for (int i = 0; i < 8; ++i) { b[i] = blo[i]; b[8 + i] = bhi[i]; }

#pragma unroll
            for (int t = 0; t < TM; ++t) {
                const int m0 = (tileBase + t) * 16;      // wave-uniform
                if (m0 >= M) break;
                const float* Arow = A + (size_t)(m0 + r) * lda + k0 + kk;
                const float4 a0 = *(const float4*)(Arow + g * 8);
                const float4 a1 = *(const float4*)(Arow + g * 8 + 4);
                const float4 a2 = *(const float4*)(Arow + 16 + g * 8);
                const float4 a3 = *(const float4*)(Arow + 16 + g * 8 + 4);
                __builtin_prefetch(Arow + KC, 0, 1);     // next K chunk
                v16h a;
                a[0]  = (_Float16)a0.x; a[1]  = (_Float16)a0.y;
                a[2]  = (_Float16)a0.z; a[3]  = (_Float16)a0.w;
                a[4]  = (_Float16)a1.x; a[5]  = (_Float16)a1.y;
                a[6]  = (_Float16)a1.z; a[7]  = (_Float16)a1.w;
                a[8]  = (_Float16)a2.x; a[9]  = (_Float16)a2.y;
                a[10] = (_Float16)a2.z; a[11] = (_Float16)a2.w;
                a[12] = (_Float16)a3.x; a[13] = (_Float16)a3.y;
                a[14] = (_Float16)a3.z; a[15] = (_Float16)a3.w;
                acc[t] = __builtin_amdgcn_wmma_f32_16x16x32_f16(
                             false, a, false, b, (short)0, acc[t], false, false);
            }
        }
        __syncthreads();   // LDS reused next chunk
    }

    const int n = n0 + r;
    const float bv = bias ? bias[n] : 0.0f;
#pragma unroll
    for (int t = 0; t < TM; ++t) {
        const int m0 = (tileBase + t) * 16;
        if (m0 >= M) break;
#pragma unroll
        for (int v = 0; v < 8; ++v) {
            float val = acc[t][v] + bv;
            if (do_act) val = LREL(val, slope);
            C[(size_t)(m0 + v + 8 * g) * ldc + n] = val;
        }
    }
}

// ---------------------------------------------------------------------------
// Tiny-K encoders: num_prop (K=5) -> x[:,64:96], cat_prop (K=3) -> x[:,96:128]
// ---------------------------------------------------------------------------
__global__ void enc_small(const float* __restrict__ np_, const float* __restrict__ cp,
                          const float* __restrict__ Wnp, const float* __restrict__ bnp,
                          const float* __restrict__ Wcp, const float* __restrict__ bcp,
                          float* __restrict__ x, int Nn)
{
    const int idx  = blockIdx.x * blockDim.x + threadIdx.x;
    const int node = idx >> 5;
    const int c    = idx & 31;
    if (node >= Nn) return;
    float s = bnp[c];
#pragma unroll
    for (int k = 0; k < 5; ++k) s += np_[(size_t)node * 5 + k] * Wnp[k * 32 + c];
    x[(size_t)node * 128 + 64 + c] = LREL(s, 0.01f);
    float s2 = bcp[c];
#pragma unroll
    for (int k = 0; k < 3; ++k) s2 += cp[(size_t)node * 3 + k] * Wcp[k * 32 + c];
    x[(size_t)node * 128 + 96 + c] = LREL(s2, 0.01f);
}

// al_s[n,h] = <h[n,h,:], a_src[h,:]>, al_d likewise
__global__ void att_coef(const float* __restrict__ h,
                         const float* __restrict__ asrc, const float* __restrict__ adst,
                         float* __restrict__ al_s, float* __restrict__ al_d,
                         int Nn, int H, int Cc)
{
    const int idx = blockIdx.x * blockDim.x + threadIdx.x;
    if (idx >= Nn * H) return;
    const int node = idx / H;
    const int hh   = idx - node * H;
    const float* hp = h + (size_t)node * 128 + hh * Cc;
    float ss = 0.0f, sd = 0.0f;
    for (int c = 0; c < Cc; ++c) {
        const float v = hp[c];
        ss += v * asrc[hh * Cc + c];
        sd += v * adst[hh * Cc + c];
    }
    al_s[idx] = ss;
    al_d[idx] = sd;
}

// out <- bias (accumulator seed), m <- -FLT_MAX, den <- 0
__global__ void gat_init(float* __restrict__ m, float* __restrict__ den,
                         float* __restrict__ out, const float* __restrict__ bias,
                         int Nn, int H)
{
    const int i = blockIdx.x * blockDim.x + threadIdx.x;
    if (i >= Nn * 128) return;
    const int f = i & 127;
    out[i] = bias[f];
    if (f < H) {
        const int j = (i >> 7) * H + f;
        m[j]   = -3.402823466e38f;
        den[j] = 0.0f;
    }
}

__device__ __forceinline__ void atomicMaxF(float* addr, float v)
{
    if (v >= 0.0f) atomicMax((int*)addr, __float_as_int(v));
    else           atomicMin((unsigned int*)addr, __float_as_uint(v));
}

__device__ __forceinline__ void edge_ends(const long long* __restrict__ ei,
                                          int E_, int e, int& s, int& d)
{
    if (e < E_) { s = (int)ei[e]; d = (int)ei[(size_t)E_ + e]; }
    else        { s = d = e - E_; }                 // appended self-loop
}

__global__ void edge_max(const long long* __restrict__ ei, int E_, int EE,
                         const float* __restrict__ al_s, const float* __restrict__ al_d,
                         float* __restrict__ m, int H)
{
    const int e = blockIdx.x * blockDim.x + threadIdx.x;
    if (e >= EE) return;
    int s, d; edge_ends(ei, E_, e, s, d);
    for (int hh = 0; hh < H; ++hh) {
        float ew = al_s[s * H + hh] + al_d[d * H + hh];
        ew = LREL(ew, 0.2f);
        atomicMaxF(&m[d * H + hh], ew);
    }
}

__global__ void edge_sum(const long long* __restrict__ ei, int E_, int EE,
                         const float* __restrict__ al_s, const float* __restrict__ al_d,
                         const float* __restrict__ m, float* __restrict__ den, int H)
{
    const int e = blockIdx.x * blockDim.x + threadIdx.x;
    if (e >= EE) return;
    int s, d; edge_ends(ei, E_, e, s, d);
    for (int hh = 0; hh < H; ++hh) {
        float ew = al_s[s * H + hh] + al_d[d * H + hh];
        ew = LREL(ew, 0.2f);
        atomicAdd(&den[d * H + hh], __expf(ew - m[d * H + hh]));
    }
}

// one wave per edge; lane handles features 4*lane..4*lane+3 of 128
__global__ void edge_agg(const long long* __restrict__ ei, int E_, int EE,
                         const float* __restrict__ al_s, const float* __restrict__ al_d,
                         const float* __restrict__ m, const float* __restrict__ den,
                         const float* __restrict__ h, float* __restrict__ out, int H)
{
    const int gtid = blockIdx.x * blockDim.x + threadIdx.x;
    const int e    = gtid >> 5;
    const int lane = gtid & 31;
    if (e >= EE) return;
    int s, d; edge_ends(ei, E_, e, s, d);
    const int f  = lane * 4;
    const int Cc = 128 / H;
    const int hh = f / Cc;
    float ew = al_s[s * H + hh] + al_d[d * H + hh];
    ew = LREL(ew, 0.2f);
    const float alpha = __expf(ew - m[d * H + hh]) / (den[d * H + hh] + 1e-16f);
    const float4 hv = *(const float4*)(h + (size_t)s * 128 + f);
    float* op = out + (size_t)d * 128 + f;
    atomicAdd(op + 0, alpha * hv.x);
    atomicAdd(op + 1, alpha * hv.y);
    atomicAdd(op + 2, alpha * hv.z);
    atomicAdd(op + 3, alpha * hv.w);
}

// logits: out[n,2] = x[n,:] @ W_o2[128,2] + b_o2
__global__ void final_linear(const float* __restrict__ x, const float* __restrict__ Wo2,
                             const float* __restrict__ bo2, float* __restrict__ out, int Nn)
{
    const int n = blockIdx.x * blockDim.x + threadIdx.x;
    if (n >= Nn) return;
    float s0 = bo2[0], s1 = bo2[1];
    const float* xp = x + (size_t)n * 128;
    for (int k = 0; k < 128; ++k) {
        const float v = xp[k];
        s0 += v * Wo2[k * 2 + 0];
        s1 += v * Wo2[k * 2 + 1];
    }
    out[n * 2 + 0] = s0;
    out[n * 2 + 1] = s1;
}

// ---------------------------------------------------------------------------
extern "C" void kernel_launch(void* const* d_in, const int* in_sizes, int n_in,
                              void* d_out, int out_size, void* d_ws, size_t ws_size,
                              hipStream_t stream)
{
    const float* des   = (const float*)d_in[0];
    const float* tweet = (const float*)d_in[1];
    const float* npr   = (const float*)d_in[2];
    const float* cpr   = (const float*)d_in[3];
    const long long* ei = (const long long*)d_in[4];   // jnp.int64 edge_index [2,E]
    const float* W_des = (const float*)d_in[5];  const float* b_des = (const float*)d_in[6];
    const float* W_tw  = (const float*)d_in[7];  const float* b_tw  = (const float*)d_in[8];
    const float* W_np  = (const float*)d_in[9];  const float* b_np  = (const float*)d_in[10];
    const float* W_cp  = (const float*)d_in[11]; const float* b_cp  = (const float*)d_in[12];
    const float* W_in  = (const float*)d_in[13]; const float* b_in  = (const float*)d_in[14];
    const float* g1_W  = (const float*)d_in[15];
    const float* g1_as = (const float*)d_in[16]; const float* g1_ad = (const float*)d_in[17];
    const float* g1_b  = (const float*)d_in[18];
    const float* g2_W  = (const float*)d_in[19];
    const float* g2_as = (const float*)d_in[20]; const float* g2_ad = (const float*)d_in[21];
    const float* g2_b  = (const float*)d_in[22];
    const float* W_o1  = (const float*)d_in[23]; const float* b_o1  = (const float*)d_in[24];
    const float* W_o2  = (const float*)d_in[25]; const float* b_o2  = (const float*)d_in[26];

    const int Nn = in_sizes[0] / 768;
    const int E_ = in_sizes[4] / 2;
    const int EE = E_ + Nn;                       // edges + self loops

    // workspace layout (floats): 3 node-feature buffers + small per-head arrays
    const size_t nFeat = (size_t)Nn * 128;
    float* b0   = (float*)d_ws;
    float* b1   = b0 + nFeat;
    float* b2   = b1 + nFeat;
    float* als  = b2 + nFeat;
    float* ald  = als + (size_t)Nn * 4;
    float* mbuf = ald + (size_t)Nn * 4;
    float* den  = mbuf + (size_t)Nn * 4;
    (void)ws_size; (void)n_in; (void)out_size;

    const int mtiles   = (Nn + 15) / 16;
    const int mblocks  = (mtiles + 4 * TM - 1) / (4 * TM);   // 4 waves x TM tiles / block
    const dim3 blkG(32, 4);
    const dim3 grdEnc(mblocks, 2);                // Ncols=32
    const dim3 grd128(mblocks, 8);                // Ncols=128
    const int T = 256;

    // ---- encoders: x(b0)[N,128] = [leaky(des@Wd) | leaky(tw@Wt) | leaky(np@Wn) | leaky(cp@Wc)]
    gemm_wmma_f16<<<grdEnc, blkG, 0, stream>>>(des,   768, W_des, 32, b_des, b0 + 0,  128, Nn, 768, 32, 0.01f, 1);
    gemm_wmma_f16<<<grdEnc, blkG, 0, stream>>>(tweet, 768, W_tw,  32, b_tw,  b0 + 32, 128, Nn, 768, 32, 0.01f, 1);
    enc_small<<<((size_t)Nn * 32 + T - 1) / T, T, 0, stream>>>(npr, cpr, W_np, b_np, W_cp, b_cp, b0, Nn);

    // ---- x1(b1) = leaky(x @ W_in + b_in)
    gemm_wmma_f16<<<grd128, blkG, 0, stream>>>(b0, 128, W_in, 128, b_in, b1, 128, Nn, 128, 128, 0.01f, 1);

    // ---- GAT layer runner
    auto run_gat = [&](const float* xin, const float* Wg, const float* as, const float* ad,
                       const float* bg, float* hbuf, float* obuf, int H) {
        const int Cc = 128 / H;
        gemm_wmma_f16<<<grd128, blkG, 0, stream>>>(xin, 128, Wg, 128, nullptr, hbuf, 128, Nn, 128, 128, 0.0f, 0);
        att_coef<<<((size_t)Nn * H + T - 1) / T, T, 0, stream>>>(hbuf, as, ad, als, ald, Nn, H, Cc);
        gat_init<<<(nFeat + T - 1) / T, T, 0, stream>>>(mbuf, den, obuf, bg, Nn, H);
        edge_max<<<(EE + T - 1) / T, T, 0, stream>>>(ei, E_, EE, als, ald, mbuf, H);
        edge_sum<<<(EE + T - 1) / T, T, 0, stream>>>(ei, E_, EE, als, ald, mbuf, den, H);
        edge_agg<<<((size_t)EE * 32 + T - 1) / T, T, 0, stream>>>(ei, E_, EE, als, ald, mbuf, den, hbuf, obuf, H);
    };

    run_gat(b1, g1_W, g1_as, g1_ad, g1_b, /*h*/ b0, /*out*/ b2, 4);   // heads=4, C=32, concat
    run_gat(b2, g2_W, g2_as, g2_ad, g2_b, /*h*/ b1, /*out*/ b0, 1);   // heads=1, C=128

    // ---- x3(b2) = leaky(out2 @ W_o1 + b_o1); logits = x3 @ W_o2 + b_o2
    gemm_wmma_f16<<<grd128, blkG, 0, stream>>>(b0, 128, W_o1, 128, b_o1, b2, 128, Nn, 128, 128, 0.01f, 1);
    final_linear<<<(Nn + T - 1) / T, T, 0, stream>>>(b2, W_o2, b_o2, (float*)d_out, Nn);
}
